// VGG_Cifar10_74242804678951
// MI455X (gfx1250) — 8-bit matrix operand order repaired and checked by reading, compiled, not run on hardware
//
#include <hip/hip_runtime.h>
#include <stdint.h>

typedef int v8i __attribute__((ext_vector_type(8)));

#define BN_EPS 1e-5f

__global__ void binarize_weights_kernel(const float* __restrict__ w,
                                        int8_t* __restrict__ out,
                                        int Cin, long total) {
    long idx = (long)blockIdx.x * blockDim.x + threadIdx.x;
    if (idx >= total) return;
    int K9  = Cin * 9;
    int co  = (int)(idx / K9);
    int rem = (int)(idx - (long)co * K9);
    int ci  = rem / 9;
    int tap = rem - ci * 9;
    out[(long)co * K9 + (long)tap * Cin + ci] = (w[idx] >= 0.0f) ? (int8_t)1 : (int8_t)-1;
}

__global__ void conv1_kernel(const float* __restrict__ x,
                             const float* __restrict__ w1,
                             const float* __restrict__ g, const float* __restrict__ b,
                             const float* __restrict__ m, const float* __restrict__ v,
                             int8_t* __restrict__ act_out) {
    int p  = blockIdx.x;
    int n  = p >> 10;
    int y  = (p >> 5) & 31;
    int xx = p & 31;

    __shared__ float patch[27];
    int t = threadIdx.x;
    if (t < 27) {
        int ci  = t / 9;
        int tap = t - ci * 9;
        int ky  = tap / 3, kx = tap - ky * 3;
        int yy = y + ky - 1, xw = xx + kx - 1;
        float val = 0.0f;
        if ((unsigned)yy < 32u && (unsigned)xw < 32u)
            val = x[((n * 3 + ci) * 32 + yy) * 32 + xw];
        patch[t] = val;
    }
    __syncthreads();

    int co = t;
    const float* wr = w1 + co * 27;
    float s = 0.0f;
    #pragma unroll
    for (int i = 0; i < 27; ++i) s = fmaf(wr[i], patch[i], s);
    float scale = g[co] * rsqrtf(v[co] + BN_EPS);
    float bnv   = (s - m[co]) * scale + b[co];
    act_out[(long)p * 384 + co] = (bnv >= 0.0f) ? (int8_t)1 : (int8_t)-1;
}

__device__ __forceinline__ v8i load_a_frag(const int8_t* p) {
    int2 t0 = *(const int2*)(p);
    int2 t1 = *(const int2*)(p + 8);
    int2 t2 = *(const int2*)(p + 32);
    int2 t3 = *(const int2*)(p + 40);
    v8i a;
    a[0] = t0.x; a[1] = t0.y; a[2] = t1.x; a[3] = t1.y;
    a[4] = t2.x; a[5] = t2.y; a[6] = t3.x; a[7] = t3.y;
    return a;
}

__device__ __forceinline__ v8i load_b_frag(const int8_t* act, long off, bool valid) {
    v8i b = {0, 0, 0, 0, 0, 0, 0, 0};
    if (valid) {
        const int8_t* p = act + off;
        int4 u0 = *(const int4*)(p);
        int4 u1 = *(const int4*)(p + 32);
        b[0] = u0.x; b[1] = u0.y; b[2] = u0.z; b[3] = u0.w;
        b[4] = u1.x; b[5] = u1.y; b[6] = u1.z; b[7] = u1.w;
    }
    return b;
}

template <int H, int W, int Cin, int Cout, int MODE>
__global__ void __launch_bounds__(128)
bconv_wmma_kernel(const int8_t* __restrict__ act,
                  const int8_t* __restrict__ wbin,
                  int8_t* __restrict__ out_i8,
                  float* __restrict__ out_f32,
                  const float* __restrict__ bn_g, const float* __restrict__ bn_b,
                  const float* __restrict__ bn_m, const float* __restrict__ bn_v) {
    constexpr int HW = H * W;
    constexpr int K9 = Cin * 9;

    const int lane = threadIdx.x & 31;
    const int wave = threadIdx.x >> 5;
    const int col  = lane & 15;
    const int half = lane >> 4;

    const int pt      = blockIdx.x * 4 + wave;
    const int ct      = blockIdx.y;
    const int p_base  = pt * 64;
    const int co_base = ct * 32;

    int nn[4], yy[4], xx[4];
    #pragma unroll
    for (int s = 0; s < 4; ++s) {
        int p = p_base + s * 16 + col;
        nn[s] = p / HW;
        int q = p % HW;
        yy[s] = q / W;
        xx[s] = q % W;
    }

    const int8_t* wrow0 = wbin + (long)(co_base + col) * K9 + half * 16;
    const int8_t* wrow1 = wbin + (long)(co_base + 16 + col) * K9 + half * 16;

    v8i acc[8];
    #pragma unroll
    for (int i = 0; i < 8; ++i) acc[i] = (v8i){0, 0, 0, 0, 0, 0, 0, 0};

    #pragma unroll 1
    for (int tap = 0; tap < 9; ++tap) {
        const int dy = tap / 3 - 1;
        const int dx = tap - (tap / 3) * 3 - 1;

        long boff[4];
        bool vld[4];
        #pragma unroll
        for (int s = 0; s < 4; ++s) {
            int py = yy[s] + dy, px = xx[s] + dx;
            vld[s]  = ((unsigned)py < (unsigned)H) && ((unsigned)px < (unsigned)W);
            boff[s] = ((long)(nn[s] * H + py) * W + px) * (long)Cin + half * 16;
        }
        const int8_t* a0p = wrow0 + (long)tap * Cin;
        const int8_t* a1p = wrow1 + (long)tap * Cin;

        #pragma unroll 2
        for (int cb = 0; cb < Cin; cb += 64) {
            v8i af0 = load_a_frag(a0p + cb);
            v8i af1 = load_a_frag(a1p + cb);
            v8i bf0 = load_b_frag(act, boff[0] + cb, vld[0]);
            v8i bf1 = load_b_frag(act, boff[1] + cb, vld[1]);
            v8i bf2 = load_b_frag(act, boff[2] + cb, vld[2]);
            v8i bf3 = load_b_frag(act, boff[3] + cb, vld[3]);
            acc[0] = __builtin_amdgcn_wmma_i32_16x16x64_iu8(true, af0, true, bf0, acc[0], false, false);
            acc[1] = __builtin_amdgcn_wmma_i32_16x16x64_iu8(true, af0, true, bf1, acc[1], false, false);
            acc[2] = __builtin_amdgcn_wmma_i32_16x16x64_iu8(true, af0, true, bf2, acc[2], false, false);
            acc[3] = __builtin_amdgcn_wmma_i32_16x16x64_iu8(true, af0, true, bf3, acc[3], false, false);
            acc[4] = __builtin_amdgcn_wmma_i32_16x16x64_iu8(true, af1, true, bf0, acc[4], false, false);
            acc[5] = __builtin_amdgcn_wmma_i32_16x16x64_iu8(true, af1, true, bf1, acc[5], false, false);
            acc[6] = __builtin_amdgcn_wmma_i32_16x16x64_iu8(true, af1, true, bf2, acc[6], false, false);
            acc[7] = __builtin_amdgcn_wmma_i32_16x16x64_iu8(true, af1, true, bf3, acc[7], false, false);
        }
    }

    #pragma unroll
    for (int sco = 0; sco < 2; ++sco) {
        int cb2 = co_base + sco * 16 + half * 8;
        float sc[8], sm[8], sb[8];
        #pragma unroll
        for (int r = 0; r < 8; ++r) {
            int co = cb2 + r;
            sc[r] = bn_g[co] * rsqrtf(bn_v[co] + BN_EPS);
            sm[r] = bn_m[co];
            sb[r] = bn_b[co];
        }
        #pragma unroll
        for (int sp = 0; sp < 4; ++sp) {
            const v8i& a = acc[sco * 4 + sp];
            int p = p_base + sp * 16 + col;
            if constexpr (MODE == 0) {
                union { int8_t c[8]; int2 q; } u;
                #pragma unroll
                for (int r = 0; r < 8; ++r) {
                    float f = ((float)a[r] - sm[r]) * sc[r] + sb[r];
                    u.c[r]  = (f >= 0.0f) ? (int8_t)1 : (int8_t)-1;
                }
                *(int2*)(out_i8 + (long)p * Cout + cb2) = u.q;
            } else {
                float fo[8];
                #pragma unroll
                for (int r = 0; r < 8; ++r) {
                    float f = ((float)a[r] - sm[r]) * sc[r] + sb[r];
                    fo[r]   = fminf(fmaxf(f, -1.0f), 1.0f);
                }
                float* o = out_f32 + (long)p * Cout + cb2;
                *(float4*)(o)     = make_float4(fo[0], fo[1], fo[2], fo[3]);
                *(float4*)(o + 4) = make_float4(fo[4], fo[5], fo[6], fo[7]);
            }
        }
    }
}

__global__ void pool_i8_kernel(const int8_t* __restrict__ in, int8_t* __restrict__ out,
                               int Ho, int Wo, int C, long total) {
    long idx = (long)blockIdx.x * blockDim.x + threadIdx.x;
    if (idx >= total) return;
    int c  = (int)(idx % C);
    long t = idx / C;
    int xo = (int)(t % Wo); t /= Wo;
    int yo = (int)(t % Ho);
    int n  = (int)(t / Ho);
    int Wi = Wo * 2, Hi = Ho * 2;
    long base = ((long)(n * Hi + 2 * yo) * Wi + 2 * xo) * C + c;
    int8_t a  = in[base];
    int8_t bq = in[base + C];
    int8_t cc = in[base + (long)Wi * C];
    int8_t d  = in[base + (long)Wi * C + C];
    int8_t m1 = a > bq ? a : bq;
    int8_t m2 = cc > d ? cc : d;
    out[idx] = m1 > m2 ? m1 : m2;
}

__global__ void pool6_f32_kernel(const float* __restrict__ in, float* __restrict__ out,
                                 long total) {
    long idx = (long)blockIdx.x * blockDim.x + threadIdx.x;
    if (idx >= total) return;
    int x2 = (int)(idx & 3);
    int y2 = (int)((idx >> 2) & 3);
    int c  = (int)((idx >> 4) & 511);
    int n  = (int)(idx >> 13);
    const float* p = in + ((long)(n * 8 + 2 * y2) * 8 + 2 * x2) * 512 + c;
    float m = fmaxf(fmaxf(p[0], p[512]), fmaxf(p[8 * 512], p[8 * 512 + 512]));
    out[idx] = m;
}

__global__ void classifier_kernel(const float* __restrict__ h,
                                  const float* __restrict__ wlin,
                                  float* __restrict__ out) {
    int n = blockIdx.x;
    int tid = threadIdx.x;
    __shared__ float red[256];
    __shared__ float logits[10];

    float acc[10];
    #pragma unroll
    for (int j = 0; j < 10; ++j) acc[j] = 0.0f;

    const float* hr = h + (long)n * 8192;
    for (int k = tid; k < 8192; k += 256) {
        float hv = hr[k];
        #pragma unroll
        for (int j = 0; j < 10; ++j) acc[j] = fmaf(hv, wlin[j * 8192 + k], acc[j]);
    }
    for (int j = 0; j < 10; ++j) {
        red[tid] = acc[j];
        __syncthreads();
        for (int s = 128; s > 0; s >>= 1) {
            if (tid < s) red[tid] += red[tid + s];
            __syncthreads();
        }
        if (tid == 0) logits[j] = red[0];
        __syncthreads();
    }
    if (tid == 0) {
        float mx = logits[0];
        #pragma unroll
        for (int j = 1; j < 10; ++j) mx = fmaxf(mx, logits[j]);
        float se = 0.0f;
        #pragma unroll
        for (int j = 0; j < 10; ++j) se += expf(logits[j] - mx);
        float lse = logf(se);
        #pragma unroll
        for (int j = 0; j < 10; ++j) out[n * 10 + j] = logits[j] - mx - lse;
    }
}

extern "C" void kernel_launch(void* const* d_in, const int* in_sizes, int n_in,
                              void* d_out, int out_size, void* d_ws, size_t ws_size,
                              hipStream_t stream) {
    (void)in_sizes; (void)n_in; (void)out_size; (void)ws_size;

    const float* x  = (const float*)d_in[0];
    const float* w1 = (const float*)d_in[1];
    const float* wsrc[5] = { (const float*)d_in[2], (const float*)d_in[3],
                             (const float*)d_in[4], (const float*)d_in[5],
                             (const float*)d_in[6] };
    const float* bn[6][4];
    for (int i = 0; i < 6; ++i)
        for (int j = 0; j < 4; ++j)
            bn[i][j] = (const float*)d_in[7 + i * 4 + j];
    const float* w_lin = (const float*)d_in[31];
    float* out = (float*)d_out;

    int8_t* ws8 = (int8_t*)d_ws;
    const size_t szW[5] = { (size_t)384 * 384 * 9, (size_t)768 * 384 * 9,
                            (size_t)768 * 768 * 9, (size_t)1536 * 768 * 9,
                            (size_t)512 * 1536 * 9 };
    size_t offW[5];
    size_t cur = 0;
    for (int i = 0; i < 5; ++i) { offW[i] = cur; cur += szW[i]; }
    size_t offA = cur;
    size_t offB = offA + (size_t)50331648;
    int8_t* Wb[5];
    for (int i = 0; i < 5; ++i) Wb[i] = ws8 + offW[i];
    int8_t* bufA = ws8 + offA;
    int8_t* bufB = ws8 + offB;

    const int CinL[5] = { 384, 384, 768, 768, 1536 };

    for (int i = 0; i < 5; ++i) {
        long t = (long)szW[i];
        binarize_weights_kernel<<<(unsigned)((t + 255) / 256), 256, 0, stream>>>(
            wsrc[i], Wb[i], CinL[i], t);
    }

    conv1_kernel<<<131072, 384, 0, stream>>>(x, w1, bn[0][0], bn[0][1], bn[0][2], bn[0][3], bufA);

    bconv_wmma_kernel<32, 32, 384, 384, 0><<<dim3(512, 12), 128, 0, stream>>>(
        bufA, Wb[0], bufB, nullptr, bn[1][0], bn[1][1], bn[1][2], bn[1][3]);
    { long t = 128L * 16 * 16 * 384;
      pool_i8_kernel<<<(unsigned)((t + 255) / 256), 256, 0, stream>>>(bufB, bufA, 16, 16, 384, t); }

    bconv_wmma_kernel<16, 16, 384, 768, 0><<<dim3(128, 24), 128, 0, stream>>>(
        bufA, Wb[1], bufB, nullptr, bn[2][0], bn[2][1], bn[2][2], bn[2][3]);

    bconv_wmma_kernel<16, 16, 768, 768, 0><<<dim3(128, 24), 128, 0, stream>>>(
        bufB, Wb[2], bufA, nullptr, bn[3][0], bn[3][1], bn[3][2], bn[3][3]);
    { long t = 128L * 8 * 8 * 768;
      pool_i8_kernel<<<(unsigned)((t + 255) / 256), 256, 0, stream>>>(bufA, bufB, 8, 8, 768, t); }

    bconv_wmma_kernel<8, 8, 768, 1536, 0><<<dim3(32, 48), 128, 0, stream>>>(
        bufB, Wb[3], bufA, nullptr, bn[4][0], bn[4][1], bn[4][2], bn[4][3]);

    bconv_wmma_kernel<8, 8, 1536, 512, 1><<<dim3(32, 16), 128, 0, stream>>>(
        bufA, Wb[4], nullptr, (float*)bufB, bn[5][0], bn[5][1], bn[5][2], bn[5][3]);
    { long t = 128L * 512 * 16;
      pool6_f32_kernel<<<(unsigned)((t + 255) / 256), 256, 0, stream>>>(
          (const float*)bufB, (float*)bufA, t); }

    classifier_kernel<<<128, 256, 0, stream>>>((const float*)bufA, w_lin, out);
}
